// MultiScaleRetention_42502996361908
// MI455X (gfx1250) — compile-verified, matrix operand order checked by reading, not run on hardware
//
#include <hip/hip_runtime.h>

// ---------------------------------------------------------------------------
// MultiScaleRetention for MI455X (gfx1250, wave32, WMMA bf16 16x16x32)
// B=4 T=8192 E=512 H=8 C=256 KD=64 HD=128 NC=32
// ---------------------------------------------------------------------------

#define Bc   4
#define Tc   8192
#define Ec   512
#define Hc   8
#define Cc   256
#define KDc  64
#define HDc  128
#define NCc  32
#define BTc  (Bc*Tc)       // 32768
#define E2c  (2*Ec)        // 1024

typedef __bf16 bf16_t;
typedef __attribute__((ext_vector_type(16))) __bf16 v16bf;
typedef __attribute__((ext_vector_type(8)))  __bf16 v8bf;
typedef __attribute__((ext_vector_type(8)))  float  v8f;

__device__ __forceinline__ bf16_t f2bf(float f){
  unsigned u = __builtin_bit_cast(unsigned, f);
  unsigned r = u + 0x7fffu + ((u >> 16) & 1u);   // round-to-nearest-even
  unsigned short h = (unsigned short)(r >> 16);
  return __builtin_bit_cast(bf16_t, h);
}
__device__ __forceinline__ float bf2f(bf16_t b){
  unsigned short h = __builtin_bit_cast(unsigned short, b);
  unsigned u = ((unsigned)h) << 16;
  return __builtin_bit_cast(float, u);
}
__device__ __forceinline__ v8f v8f_zero(){
  v8f z = {0.f,0.f,0.f,0.f,0.f,0.f,0.f,0.f};
  return z;
}
__device__ __forceinline__ v8f wmma_bf16(v16bf a, v16bf b, v8f c){
  return __builtin_amdgcn_wmma_f32_16x16x32_bf16(false, a, false, b, (short)0, c,
                                                 false, false);
}

// A/B fragment layout (16x16x32 bf16, wave32):
//   row/col = lane & 15 ;  element j -> k = (j<8 ? j : j+8) + ((lane&16)?8:0)
// Caller passes pointer already offset by (k0 + koff); memory is K-contiguous,
// so the fragment is two 16-byte loads (k..k+7 and k+16..k+23).
__device__ __forceinline__ v16bf load_frag16(const bf16_t* pr){
  const v8bf* q = (const v8bf*)pr;
  v8bf lo = q[0];
  v8bf hi = q[2];
  v16bf r;
  #pragma unroll
  for (int j=0;j<8;j++){ r[j] = lo[j]; r[8+j] = hi[j]; }
  return r;
}

// ---------------------------------------------------------------------------
// Converts
// ---------------------------------------------------------------------------
__global__ void cvt_bf16_kernel(const float* __restrict__ in,
                                bf16_t* __restrict__ out, long n){
  long i = (long)blockIdx.x * blockDim.x + threadIdx.x;
  if (i < n) out[i] = f2bf(in[i]);
}

// in: K x N row-major f32 ; out: N x K row-major bf16 (K-contiguous for B frags)
__global__ void cvt_wT_kernel(const float* __restrict__ in,
                              bf16_t* __restrict__ out, int K, int N){
  long i = (long)blockIdx.x * blockDim.x + threadIdx.x;
  if (i < (long)K*N){
    int k = (int)(i / N), n = (int)(i % N);
    out[(long)n*K + k] = f2bf(in[i]);
  }
}

// ---------------------------------------------------------------------------
// Projection GEMM: Y[m,n] = sum_k Xb[m,k] * WT[n,k]
// grid.x = BT/16 row tiles; each wave owns a 16x64 strip (1 A-frag, 4 WMMAs/kstep)
// mode 0: q -> theta-shift -> qr [B,NC,H,C,KD]
// mode 1: k -> *KD^-0.5 -> theta-shift -> kr [B,NC,H,C,KD] + krT [B,NC,H,KD,C]
// mode 2: v -> vT [B,NC,H,HD,C]     mode 3: g -> [BT, 2E]
// ---------------------------------------------------------------------------
__global__ void proj_gemm_kernel(const bf16_t* __restrict__ X,
                                 const bf16_t* __restrict__ WT,
                                 bf16_t* __restrict__ out,
                                 bf16_t* __restrict__ out2,
                                 int mode,
                                 const float* __restrict__ sinT,
                                 const float* __restrict__ cosT){
  const int lane = threadIdx.x & 31;
  const int wave = threadIdx.x >> 5;
  const int ln   = lane & 15;
  const int mhi  = (lane & 16) ? 8 : 0;
  const int koff = mhi;
  const int row0 = blockIdx.x * 16;
  const bf16_t* arow = X + (long)(row0 + ln) * Ec;
  const int n0 = wave * 64 + ln;

  v8f acc[4];
  #pragma unroll
  for (int t=0;t<4;t++) acc[t] = v8f_zero();

  for (int k0 = 0; k0 < Ec; k0 += 32){
    v16bf a = load_frag16(arow + k0 + koff);
    #pragma unroll
    for (int t=0;t<4;t++){
      v16bf b = load_frag16(WT + (long)(n0 + t*16) * Ec + k0 + koff);
      acc[t] = wmma_bf16(a, b, acc[t]);
    }
  }

  #pragma unroll
  for (int t=0;t<4;t++){
    const int n = n0 + t*16;
    #pragma unroll
    for (int r=0;r<8;r++){
      const int m  = row0 + r + mhi;
      const int bb = m / Tc;
      const int tt = m % Tc;
      const int nc = tt >> 8;
      const int c  = tt & (Cc-1);
      float val = acc[t][r];
      if (mode <= 1){
        if (mode == 1) val *= 0.125f;                 // KD^-0.5
        float p   = __shfl_xor(val, 1, 32);           // partner col n^1
        float rot = (n & 1) ? p : -p;                 // rotate_every_two
        const int   kd = n & (KDc-1);
        const float cv = cosT[(long)tt*KDc + kd];
        const float sv = sinT[(long)tt*KDc + kd];
        const float o  = val * cv + rot * sv;
        const int hh = n >> 6;
        const long off = ((((long)bb*NCc + nc)*Hc + hh)*Cc + c)*(long)KDc + kd;
        out[off] = f2bf(o);
        if (mode == 1){
          const long off2 = ((((long)bb*NCc + nc)*Hc + hh)*(long)KDc + kd)*Cc + c;
          out2[off2] = f2bf(o);
        }
      } else if (mode == 2){
        const int hh = n >> 7;
        const int hd = n & (HDc-1);
        const long off = ((((long)bb*NCc + nc)*Hc + hh)*(long)HDc + hd)*Cc + c;
        out[off] = f2bf(val);
      } else {
        out[(long)m*E2c + n] = f2bf(val);
      }
    }
  }
}

// ---------------------------------------------------------------------------
// Per-chunk inner attention: qk = (qr @ kr^T) * mask ; inner_scale = clip(sum|qk|,1)
// io_raw = qk @ v   (row-scales are folded into the final combine kernel)
// One block per (b,nc,h); 4 waves; each wave owns 4 x 16-row strips.
// kr tile (256x64 bf16 = 32KB) is async-staged to LDS once per block
// (global_load_async_to_lds_b128 / ASYNCcnt); qk strip is transposed
// lane-layout D->A through per-wave LDS buffers (bf16).
// ---------------------------------------------------------------------------
#define QSP 256
__global__ void chunk_inner_kernel(const bf16_t* __restrict__ qr,
                                   const bf16_t* __restrict__ kr,
                                   const bf16_t* __restrict__ vT,
                                   const float* __restrict__ mask,
                                   float* __restrict__ io_raw,
                                   float* __restrict__ inner_scale){
  __shared__ __align__(16) bf16_t kts[Cc*KDc];      // staged kr tile, 32KB
  __shared__ __align__(16) bf16_t qs[4 * 16 * QSP]; // qk transpose, 32KB
  const int lane = threadIdx.x & 31;
  const int wave = threadIdx.x >> 5;       // 4 waves
  const int ln   = lane & 15;
  const int mhi  = (lane & 16) ? 8 : 0;
  const int koff = mhi;
  const int blk  = blockIdx.x;             // ((b*NC)+nc)*H + h
  const int h    = blk & (Hc-1);
  const bf16_t* qt = qr + (long)blk * (Cc*KDc);
  const bf16_t* kt = kr + (long)blk * (Cc*KDc);
  const bf16_t* vt = vT + (long)blk * (Cc*HDc);
  float* iob = io_raw      + (long)blk * (Cc*HDc);
  float* isb = inner_scale + (long)blk * Cc;
  bf16_t* myqs = qs + wave * (16*QSP);

  // ---- async stage kr tile into LDS (2048 x 16B segments) ----
  {
    const unsigned ldsbase = (unsigned)(uintptr_t)(void*)&kts[0];
    const char* gsrc = (const char*)kt;
    for (int seg = threadIdx.x; seg < (Cc*KDc*2)/16; seg += 128){
      unsigned la = ldsbase + seg*16;
      const char* ga = gsrc + seg*16;
      asm volatile("global_load_async_to_lds_b128 %0, %1, off"
                   :: "v"(la), "v"(ga) : "memory");
    }
    asm volatile("s_wait_asynccnt 0" ::: "memory");
    __syncthreads();
  }

  for (int s = wave; s < 16; s += 4){
    const int m0 = s * 16;
    float rowabs[8];
    #pragma unroll
    for (int r=0;r<8;r++) rowabs[r] = 0.f;

    // A-frags for this strip depend only on ks -> load once
    v16bf aq[2];
    #pragma unroll
    for (int ks=0; ks<2; ks++)
      aq[ks] = load_frag16(qt + (long)(m0 + ln)*KDc + ks*32 + koff);

    // qk strip: 16 x 256, K = 64 ; B = kr^T from LDS
    for (int nt=0; nt<16; nt++){
      v8f d = v8f_zero();
      #pragma unroll
      for (int ks=0; ks<2; ks++){
        v16bf b = load_frag16(kts + (nt*16 + ln)*KDc + ks*32 + koff);
        d = wmma_bf16(aq[ks], b, d);
      }
      #pragma unroll
      for (int r=0;r<8;r++){
        const int m = m0 + r + mhi;
        const int n = nt*16 + ln;
        float val = d[r] * mask[((long)h*Cc + m)*Cc + n];
        rowabs[r] += fabsf(val);
        myqs[(r+mhi)*QSP + n] = f2bf(val);
      }
    }
    // reduce |qk| row sums across the 16 lanes of each half-wave
    #pragma unroll
    for (int off=1; off<16; off<<=1){
      #pragma unroll
      for (int r=0;r<8;r++) rowabs[r] += __shfl_xor(rowabs[r], off, 32);
    }
    if (ln == 0){
      #pragma unroll
      for (int r=0;r<8;r++) isb[m0 + r + mhi] = fmaxf(rowabs[r], 1.0f);
    }
    asm volatile("s_wait_dscnt 0" ::: "memory");

    // io_raw strip: 16 x 128, K = 256 (A from LDS; A-frags hoisted)
    v16bf a8[8];
    #pragma unroll
    for (int ks=0; ks<8; ks++)
      a8[ks] = load_frag16(myqs + ln*QSP + ks*32 + koff);
    for (int nt=0; nt<8; nt++){
      v8f d = v8f_zero();
      #pragma unroll
      for (int ks=0; ks<8; ks++){
        v16bf b = load_frag16(vt + (long)(nt*16 + ln)*Cc + ks*32 + koff);
        d = wmma_bf16(a8[ks], b, d);
      }
      #pragma unroll
      for (int r=0;r<8;r++){
        iob[(long)(m0 + r + mhi)*HDc + nt*16 + ln] = d[r];
      }
    }
  }
}

// ---------------------------------------------------------------------------
// kv chunk contribution: kv = kr^T @ (v * vid)  -> [KD, HD] f32 per (b,nc,h)
// ---------------------------------------------------------------------------
__global__ void kv_chunk_kernel(const bf16_t* __restrict__ krT,
                                const bf16_t* __restrict__ vT,
                                const float* __restrict__ vid,
                                float* __restrict__ kv_chunk){
  const int lane = threadIdx.x & 31;
  const int wave = threadIdx.x >> 5;      // 8 waves
  const int ln   = lane & 15;
  const int mhi  = (lane & 16) ? 8 : 0;
  const int koff = mhi;
  const int blk  = blockIdx.x;
  const int h    = blk & (Hc-1);
  const bf16_t* kt = krT + (long)blk * (KDc*Cc);
  const bf16_t* vt = vT  + (long)blk * (Cc*HDc);
  float* ob = kv_chunk + (long)blk * (KDc*HDc);
  const int mt  = wave >> 1;              // 4 M tiles
  const int ntb = (wave & 1) * 4;         // 8 N tiles split in two

  // A-frags depend only on (mt, ks) -> hoist out of the N-tile loop
  v16bf a8[8];
  #pragma unroll
  for (int ks=0; ks<8; ks++)
    a8[ks] = load_frag16(kt + (long)(mt*16 + ln)*Cc + ks*32 + koff);

  for (int nti=0; nti<4; nti++){
    const int nt = ntb + nti;
    v8f d = v8f_zero();
    for (int ks=0; ks<8; ks++){
      v16bf b = load_frag16(vt + (long)(nt*16 + ln)*Cc + ks*32 + koff);
      #pragma unroll
      for (int j=0;j<16;j++){
        const int kk = ks*32 + koff + ((j<8)? j : (j+8));
        b[j] = f2bf(bf2f(b[j]) * vid[h*Cc + kk]);
      }
      d = wmma_bf16(a8[ks], b, d);
    }
    #pragma unroll
    for (int r=0;r<8;r++)
      ob[(long)(mt*16 + r + mhi)*HDc + nt*16 + ln] = d[r];
  }
}

// ---------------------------------------------------------------------------
// Sequential scan over chunks: state' = state*cross_decay + kv_i
// Emits pre-update state (transposed bf16 [HD][KD]) and cross_scale per chunk.
// grid = B*H (sequential dependence); tiny FLOPs, f32 in LDS.
// ---------------------------------------------------------------------------
__global__ void scan_kernel(const float* __restrict__ kv_chunk,
                            bf16_t* __restrict__ kv_preT,
                            float* __restrict__ cross_scale,
                            const float* __restrict__ cross_decay){
  __shared__ float st[KDc*HDc];
  __shared__ float colsum[HDc];
  const int bh = blockIdx.x;
  const int b = bh / Hc, h = bh % Hc;
  const float cd = cross_decay[h];
  const int tid = threadIdx.x;            // 128 threads
  for (int i = tid; i < KDc*HDc; i += 128) st[i] = 0.f;
  __syncthreads();
  for (int nc = 0; nc < NCc; nc++){
    const long blk = (((long)b*NCc + nc)*Hc + h);
    for (int i = tid; i < KDc*HDc; i += 128){
      const int kd = i / HDc, hd = i % HDc;
      kv_preT[blk*(KDc*HDc) + (long)hd*KDc + kd] = f2bf(st[i]);
    }
    if (tid < HDc){
      float s = 0.f;
      for (int kd = 0; kd < KDc; kd++) s += fabsf(st[kd*HDc + tid]);
      colsum[tid] = s;
    }
    __syncthreads();
    if (tid == 0){
      float mx = 1.0f;
      for (int i = 0; i < HDc; i++) mx = fmaxf(mx, colsum[i]);
      cross_scale[blk] = mx;
    }
    __syncthreads();
    const float* kc = kv_chunk + blk*(KDc*HDc);
    for (int i = tid; i < KDc*HDc; i += 128) st[i] = st[i]*cd + kc[i];
    __syncthreads();
  }
}

// ---------------------------------------------------------------------------
// cross_output + combine + RMS-norm + silu(g) gate -> att bf16 [BT, 2E]
//   out = (io_raw + qid[row] * (qr @ kv_pre)) / max(inner_scale[row], cross_scale)
// ---------------------------------------------------------------------------
__global__ void cross_combine_kernel(const bf16_t* __restrict__ qr,
                                     const bf16_t* __restrict__ kv_preT,
                                     const float* __restrict__ io_raw,
                                     const float* __restrict__ inner_scale,
                                     const float* __restrict__ cross_scale,
                                     const float* __restrict__ qid,
                                     const bf16_t* __restrict__ g,
                                     bf16_t* __restrict__ att){
  const int lane = threadIdx.x & 31;
  const int wave = threadIdx.x >> 5;       // 4 waves
  const int ln   = lane & 15;
  const int mhi  = (lane & 16) ? 8 : 0;
  const int koff = mhi;
  const int blk  = blockIdx.x;             // ((b*NC)+nc)*H + h
  const int h    = blk & (Hc-1);
  const int bn   = blk >> 3;
  const int b    = bn / NCc;
  const int nc   = bn % NCc;
  const bf16_t* qt  = qr      + (long)blk * (Cc*KDc);
  const bf16_t* kvp = kv_preT + (long)blk * (KDc*HDc);
  const float* iob  = io_raw      + (long)blk * (Cc*HDc);
  const float* isb  = inner_scale + (long)blk * Cc;
  const float  csc  = cross_scale[blk];

  for (int s = wave; s < 16; s += 4){
    const int m0 = s*16;
    float outv[8][8];
    float ss[8], isc[8], qd[8];
    #pragma unroll
    for (int r=0;r<8;r++){
      ss[r] = 0.f;
      const int m = m0 + r + mhi;
      isc[r] = isb[m];
      qd[r]  = qid[h*Cc + m];
    }
    // A-frags for this strip depend only on ks -> load once
    v16bf aq[2];
    #pragma unroll
    for (int ks=0; ks<2; ks++)
      aq[ks] = load_frag16(qt + (long)(m0 + ln)*KDc + ks*32 + koff);

    for (int nt=0; nt<8; nt++){
      v8f d = v8f_zero();
      #pragma unroll
      for (int ks=0; ks<2; ks++){
        v16bf bb = load_frag16(kvp + (long)(nt*16 + ln)*KDc + ks*32 + koff);
        d = wmma_bf16(aq[ks], bb, d);
      }
      #pragma unroll
      for (int r=0;r<8;r++){
        const int m = m0 + r + mhi;
        const int n = nt*16 + ln;
        const float allsc = fmaxf(isc[r], csc);
        const float val = (iob[(long)m*HDc + n] + qd[r]*d[r]) / allsc;
        outv[nt][r] = val;
        ss[r] += val*val;
      }
    }
    #pragma unroll
    for (int off=1; off<16; off<<=1){
      #pragma unroll
      for (int r=0;r<8;r++) ss[r] += __shfl_xor(ss[r], off, 32);
    }
    float rms[8];
    #pragma unroll
    for (int r=0;r<8;r++) rms[r] = rsqrtf(ss[r] * (1.0f/HDc) + 1e-6f);
    for (int nt=0; nt<8; nt++){
      #pragma unroll
      for (int r=0;r<8;r++){
        const int m = m0 + r + mhi;
        const int n = nt*16 + ln;
        const int t = nc*Cc + m;
        const long row = ((long)b*Tc + t) * E2c;
        const int col  = h*HDc + n;
        const float gv = bf2f(g[row + col]);
        const float sg = gv / (1.f + __expf(-gv));     // silu
        att[row + col] = f2bf(sg * outv[nt][r] * rms[r]);
      }
    }
  }
}

// ---------------------------------------------------------------------------
// Final GEMM: out = att (BT x 1024) @ Wo (1024 x 512), f32 output
// ---------------------------------------------------------------------------
__global__ void out_gemm_kernel(const bf16_t* __restrict__ A,
                                const bf16_t* __restrict__ WoT,
                                float* __restrict__ out){
  const int lane = threadIdx.x & 31;
  const int wave = threadIdx.x >> 5;       // 8 waves
  const int ln   = lane & 15;
  const int mhi  = (lane & 16) ? 8 : 0;
  const int koff = mhi;
  const int row0 = blockIdx.x * 16;
  const bf16_t* arow = A + (long)(row0 + ln) * E2c;
  const int n0 = wave*64 + ln;
  v8f acc[4];
  #pragma unroll
  for (int t=0;t<4;t++) acc[t] = v8f_zero();
  for (int k0=0; k0<E2c; k0+=32){
    v16bf a = load_frag16(arow + k0 + koff);
    #pragma unroll
    for (int t=0;t<4;t++){
      v16bf b = load_frag16(WoT + (long)(n0 + t*16)*E2c + k0 + koff);
      acc[t] = wmma_bf16(a, b, acc[t]);
    }
  }
  #pragma unroll
  for (int t=0;t<4;t++){
    #pragma unroll
    for (int r=0;r<8;r++)
      out[(long)(row0 + r + mhi)*Ec + n0 + t*16] = acc[t][r];
  }
}

// ---------------------------------------------------------------------------
// Host launcher
// ---------------------------------------------------------------------------
extern "C" void kernel_launch(void* const* d_in, const int* in_sizes, int n_in,
                              void* d_out, int out_size, void* d_ws, size_t ws_size,
                              hipStream_t stream){
  (void)in_sizes; (void)n_in; (void)out_size; (void)ws_size;
  const float* x   = (const float*)d_in[0];
  const float* sn  = (const float*)d_in[1];
  const float* cs  = (const float*)d_in[2];
  const float* msk = (const float*)d_in[3];
  const float* cdk = (const float*)d_in[4];
  const float* qid = (const float*)d_in[5];
  const float* vid = (const float*)d_in[6];
  const float* Wq  = (const float*)d_in[7];
  const float* Wk  = (const float*)d_in[8];
  const float* Wv  = (const float*)d_in[9];
  const float* Wg  = (const float*)d_in[10];
  const float* Wo  = (const float*)d_in[11];
  float* out = (float*)d_out;

  char* p = (char*)d_ws;
  auto take = [&](size_t bytes) -> char* {
    char* r = p;
    p += (bytes + 255) & ~(size_t)255;
    return r;
  };
  bf16_t* xb   = (bf16_t*)take((size_t)BTc*Ec*2);
  bf16_t* WqT  = (bf16_t*)take((size_t)Ec*Ec*2);
  bf16_t* WkT  = (bf16_t*)take((size_t)Ec*Ec*2);
  bf16_t* WvT  = (bf16_t*)take((size_t)Ec*E2c*2);
  bf16_t* WgT  = (bf16_t*)take((size_t)Ec*E2c*2);
  bf16_t* WoT  = (bf16_t*)take((size_t)E2c*Ec*2);
  bf16_t* qrw  = (bf16_t*)take((size_t)BTc*Ec*2);        // [B,NC,H,C,KD]
  bf16_t* krw  = (bf16_t*)take((size_t)BTc*Ec*2);        // [B,NC,H,C,KD]
  bf16_t* krTw = (bf16_t*)take((size_t)BTc*Ec*2);        // [B,NC,H,KD,C]
  bf16_t* vTw  = (bf16_t*)take((size_t)BTc*E2c*2);       // [B,NC,H,HD,C]
  bf16_t* gw   = (bf16_t*)take((size_t)BTc*E2c*2);       // [BT, 2E]
  float*  iorw = (float*)take((size_t)BTc*E2c*4);        // [B,NC,H,C,HD] f32
  float*  iscw = (float*)take((size_t)Bc*NCc*Hc*Cc*4);
  float*  kvcw = (float*)take((size_t)Bc*NCc*Hc*KDc*HDc*4);
  bf16_t* kvpw = (bf16_t*)take((size_t)Bc*NCc*Hc*KDc*HDc*2); // transposed [HD,KD]
  float*  cscw = (float*)take((size_t)Bc*NCc*Hc*4);
  bf16_t* attw = (bf16_t*)take((size_t)BTc*E2c*2);

  {
    long n = (long)BTc*Ec;
    cvt_bf16_kernel<<<(int)((n+255)/256), 256, 0, stream>>>(x, xb, n);
  }
  cvt_wT_kernel<<<(Ec*Ec  + 255)/256, 256, 0, stream>>>(Wq, WqT, Ec, Ec);
  cvt_wT_kernel<<<(Ec*Ec  + 255)/256, 256, 0, stream>>>(Wk, WkT, Ec, Ec);
  cvt_wT_kernel<<<(Ec*E2c + 255)/256, 256, 0, stream>>>(Wv, WvT, Ec, E2c);
  cvt_wT_kernel<<<(Ec*E2c + 255)/256, 256, 0, stream>>>(Wg, WgT, Ec, E2c);
  cvt_wT_kernel<<<(E2c*Ec + 255)/256, 256, 0, stream>>>(Wo, WoT, E2c, Ec);

  const int rowBlocks = BTc/16;   // 2048
  proj_gemm_kernel<<<rowBlocks, 256, 0, stream>>>(xb, WqT, qrw, nullptr, 0, sn, cs);
  proj_gemm_kernel<<<rowBlocks, 256, 0, stream>>>(xb, WkT, krw, krTw,   1, sn, cs);
  proj_gemm_kernel<<<rowBlocks, 512, 0, stream>>>(xb, WvT, vTw, nullptr, 2, nullptr, nullptr);
  proj_gemm_kernel<<<rowBlocks, 512, 0, stream>>>(xb, WgT, gw,  nullptr, 3, nullptr, nullptr);

  const int chunks = Bc*NCc*Hc;   // 1024
  chunk_inner_kernel<<<chunks, 128, 0, stream>>>(qrw, krw, vTw, msk, iorw, iscw);
  kv_chunk_kernel  <<<chunks, 256, 0, stream>>>(krTw, vTw, vid, kvcw);
  scan_kernel      <<<Bc*Hc,  128, 0, stream>>>(kvcw, kvpw, cscw, cdk);
  cross_combine_kernel<<<chunks, 128, 0, stream>>>(qrw, kvpw, iorw, iscw, cscw,
                                                   qid, gw, attw);
  out_gemm_kernel<<<rowBlocks, 256, 0, stream>>>(attw, WoT, out);
}
